// LSTM_63591285785319
// MI455X (gfx1250) — matrix operand order repaired and checked by reading, not compiled, not run on hardware
//
#include <hip/hip_runtime.h>
#include <hip/hip_bf16.h>
#include <math.h>

#define B_ 64
#define S_ 1024
#define D_ 1024
#define H_ 1024

typedef __attribute__((ext_vector_type(16))) __bf16 v16bf;
typedef __attribute__((ext_vector_type(8)))  float  v8f;

union FragU { uint4 q[2]; v16bf v; };

__device__ __forceinline__ unsigned short f2bf(float f) {
    unsigned int u = __float_as_uint(f);
    unsigned int r = (u + 0x7FFFu + ((u >> 16) & 1u)) >> 16;  // round-to-nearest-even
    return (unsigned short)r;
}

__device__ __forceinline__ v16bf load_frag(const unsigned short* p0, const unsigned short* p1) {
    FragU f;
    f.q[0] = *reinterpret_cast<const uint4*>(p0);
    f.q[1] = *reinterpret_cast<const uint4*>(p1);
    return f.v;
}

// ---------------- conversion / setup kernels ----------------

__global__ void k_cast_bf16(const float* __restrict__ src, unsigned short* __restrict__ dst, int n) {
    int stride = gridDim.x * blockDim.x;
    for (int i = blockIdx.x * blockDim.x + threadIdx.x; i < n; i += stride)
        dst[i] = f2bf(src[i]);
}

// src: [4][1024][1024] f32 (k,d,h) -> dst: [4][1024][1024] bf16 (k,h,d)
__global__ void k_transpose_bf16(const float* __restrict__ src, unsigned short* __restrict__ dst) {
    int stride = gridDim.x * blockDim.x;
    const int n = 4 * D_ * H_;
    for (int i = blockIdx.x * blockDim.x + threadIdx.x; i < n; i += stride) {
        int k = i >> 20;
        int r = i & ((D_ * H_) - 1);
        int d = r >> 10;
        int h = r & (H_ - 1);
        dst[((unsigned)k * H_ + h) * D_ + d] = f2bf(src[i]);
    }
}

__global__ void k_bias_init(const float* __restrict__ bW, const float* __restrict__ bU,
                            float* __restrict__ bias, unsigned short* __restrict__ h0,
                            unsigned* __restrict__ bar) {
    int stride = gridDim.x * blockDim.x;
    int i0 = blockIdx.x * blockDim.x + threadIdx.x;
    if (i0 == 0) *bar = 0u;
    for (int j = i0; j < 4 * H_; j += stride) bias[j] = bW[j] + bU[j];
    for (int j = i0; j < B_ * H_; j += stride) h0[j] = 0;
}

// ---------------- phase 1: xproj[t][k][b][h] = x[b][t][:] . W[k][:][h] ----------------
// grid: (H/64, B/16, S*4); block: 128 (4 waves, each a 16x16 N-tile)

__global__ void __launch_bounds__(128) k_xproj(const unsigned short* __restrict__ xbf,
                                               const unsigned short* __restrict__ Wt,
                                               float* __restrict__ xproj) {
    const int lane = threadIdx.x & 31;
    const int wave = threadIdx.x >> 5;
    const int t  = blockIdx.z >> 2;
    const int k  = blockIdx.z & 3;
    const int m0 = blockIdx.y << 4;                   // batch base
    const int n0 = (blockIdx.x << 6) + (wave << 4);   // hidden base

    const int l15 = lane & 15;
    const int hi  = lane >> 4;
    const unsigned aBase = ((unsigned)(m0 + l15) * S_ + t) * D_;  // x[b][t][*]
    const int aoff = hi ? 8 : 0;
    const unsigned nb = n0 + l15;
    const unsigned bBase = ((unsigned)k * H_ + nb) * D_;          // Wt[k][n][*]
    const int boff = hi ? 8 : 0;

    v8f acc0 = {};
    v8f acc1 = {};
    for (int d0 = 0; d0 < D_; d0 += 64) {   // two independent accumulator chains
        v16bf a0 = load_frag(xbf + aBase + d0 + aoff,      xbf + aBase + d0 + aoff + 16);
        v16bf b0 = load_frag(Wt + bBase + d0 + boff,       Wt + bBase + d0 + boff + 16);
        acc0 = __builtin_amdgcn_wmma_f32_16x16x32_bf16(false, a0, false, b0,
                                                       (short)0, acc0, false, false);
        v16bf a1 = load_frag(xbf + aBase + d0 + 32 + aoff, xbf + aBase + d0 + 32 + aoff + 16);
        v16bf b1 = load_frag(Wt + bBase + d0 + 32 + boff,  Wt + bBase + d0 + 32 + boff + 16);
        acc1 = __builtin_amdgcn_wmma_f32_16x16x32_bf16(false, a1, false, b1,
                                                       (short)0, acc1, false, false);
    }
    float* outp = xproj + (size_t)(t * 4 + k) * B_ * H_;
    const int mb = 8 * hi;
#pragma unroll
    for (int j = 0; j < 8; ++j)
        outp[(unsigned)(m0 + mb + j) * H_ + nb] = acc0[j] + acc1[j];
}

// ---------------- phase 2: persistent fused LSTM scan ----------------
// grid: (H/16, B/16) = 256 blocks; block 128 = 4 waves = 4 gates of one 16x16 tile.
// Device-wide barrier (monotonic counter) between timesteps; h ping-pong buffers.
// Cell state c lives in registers (2 elements/thread) for the entire scan.

__global__ void __launch_bounds__(128) k_lstm_scan(unsigned short* __restrict__ h0,
                                                   unsigned short* __restrict__ h1,
                                                   const unsigned short* __restrict__ Ut,
                                                   const float* __restrict__ xproj, // [S][4][B][H]
                                                   const float* __restrict__ bias,  // [4][H]
                                                   float* __restrict__ out,         // [B][S][H]
                                                   float* __restrict__ hlast,
                                                   float* __restrict__ clast,
                                                   unsigned* __restrict__ bar) {
    __shared__ float gsm[4][256];
    const int lane = threadIdx.x & 31;
    const int k    = threadIdx.x >> 5;   // gate (f,i,o,g)
    const int m0   = blockIdx.y << 4;
    const int n0   = blockIdx.x << 4;

    const int l15 = lane & 15;
    const int hi  = lane >> 4;
    const unsigned aRow  = (unsigned)(m0 + l15) * H_;
    const int aoff = hi ? 8 : 0;
    const unsigned nb = n0 + l15;
    const unsigned bBase = ((unsigned)k * H_ + nb) * H_;   // Ut[k][n][*]
    const int boff = hi ? 8 : 0;
    const float bsv = bias[k * H_ + nb];
    const int mb = 8 * hi;
    const unsigned nblocks = gridDim.x * gridDim.y;

    float creg[2] = {0.0f, 0.0f};   // cell state: elements threadIdx.x, threadIdx.x+128

    for (int t = 0; t < S_; ++t) {
        const unsigned short* hp = (t & 1) ? h1 : h0;
        unsigned short*       hn = (t & 1) ? h0 : h1;
        const float* xk = xproj + ((size_t)t * 4 + k) * B_ * H_;

        // issue the x-projection tile loads early so they complete under the GEMM
        v8f xv;
#pragma unroll
        for (int j = 0; j < 8; ++j)
            xv[j] = xk[(unsigned)(m0 + mb + j) * H_ + nb];

        v8f acc0 = {};
        v8f acc1 = {};
#pragma unroll 2
        for (int d0 = 0; d0 < H_; d0 += 64) {   // two independent accumulator chains
            v16bf a0 = load_frag(hp + aRow + d0 + aoff,      hp + aRow + d0 + aoff + 16);
            v16bf b0 = load_frag(Ut + bBase + d0 + boff,     Ut + bBase + d0 + boff + 16);
            acc0 = __builtin_amdgcn_wmma_f32_16x16x32_bf16(false, a0, false, b0,
                                                           (short)0, acc0, false, false);
            v16bf a1 = load_frag(hp + aRow + d0 + 32 + aoff, hp + aRow + d0 + 32 + aoff + 16);
            v16bf b1 = load_frag(Ut + bBase + d0 + 32 + boff, Ut + bBase + d0 + 32 + boff + 16);
            acc1 = __builtin_amdgcn_wmma_f32_16x16x32_bf16(false, a1, false, b1,
                                                           (short)0, acc1, false, false);
        }

#pragma unroll
        for (int j = 0; j < 8; ++j) {
            float g = acc0[j] + acc1[j] + xv[j] + bsv;
            gsm[k][(mb + j) * 16 + l15] = 1.0f / (1.0f + __expf(-g));  // all gates sigmoid (ref!)
        }
        __syncthreads();

#pragma unroll
        for (int q = 0; q < 2; ++q) {
            int e = threadIdx.x + q * 128;
            int b = m0 + (e >> 4), h = n0 + (e & 15);
            float f  = gsm[0][e];
            float ii = gsm[1][e];
            float o  = gsm[2][e];
            float g  = gsm[3][e];
            float cn = f * creg[q] + ii * g;
            float hv = o * tanhf(cn);
            creg[q] = cn;
            out[((size_t)b * S_ + t) * H_ + h] = hv;
            hn[(unsigned)b * H_ + h] = f2bf(hv);
            if (t == S_ - 1) {
                hlast[(unsigned)b * H_ + h] = hv;
                clast[(unsigned)b * H_ + h] = cn;
            }
        }

        // ---- device-wide barrier: release writes, arrive, spin on monotonic counter ----
        __threadfence();
        __syncthreads();
        if (threadIdx.x == 0) {
            atomicAdd(bar, 1u);
            const unsigned target = nblocks * (unsigned)(t + 1);
            while (__hip_atomic_load(bar, __ATOMIC_ACQUIRE, __HIP_MEMORY_SCOPE_AGENT) < target)
                __builtin_amdgcn_s_sleep(1);
        }
        __syncthreads();
    }
}

// ---------------- host launcher ----------------

extern "C" void kernel_launch(void* const* d_in, const int* in_sizes, int n_in,
                              void* d_out, int out_size, void* d_ws, size_t ws_size,
                              hipStream_t stream) {
    (void)in_sizes; (void)n_in; (void)out_size; (void)ws_size;
    const float* input = (const float*)d_in[0];  // (B,S,D)
    const float* W     = (const float*)d_in[1];  // (4,D,H)
    const float* bW    = (const float*)d_in[2];  // (4,H)
    const float* U     = (const float*)d_in[3];  // (4,H,H)
    const float* bU    = (const float*)d_in[4];  // (4,H)
    float* out = (float*)d_out;

    char* ws = (char*)d_ws;
    size_t off = 0;
    unsigned short* xbf = (unsigned short*)(ws + off); off += (size_t)B_ * S_ * D_ * 2;
    unsigned short* Wt  = (unsigned short*)(ws + off); off += (size_t)4 * H_ * D_ * 2;
    unsigned short* Ut  = (unsigned short*)(ws + off); off += (size_t)4 * H_ * H_ * 2;
    float*          bias= (float*)(ws + off);          off += (size_t)4 * H_ * 4;
    unsigned short* h0  = (unsigned short*)(ws + off); off += (size_t)B_ * H_ * 2;
    unsigned short* h1  = (unsigned short*)(ws + off); off += (size_t)B_ * H_ * 2;
    unsigned*       bar = (unsigned*)(ws + off);       off += 256;   // keep 16B+ alignment
    float*          xproj=(float*)(ws + off);          // [S][4][B][H] f32

    k_cast_bf16<<<4096, 256, 0, stream>>>(input, xbf, B_ * S_ * D_);
    k_transpose_bf16<<<2048, 256, 0, stream>>>(W, Wt);
    k_transpose_bf16<<<2048, 256, 0, stream>>>(U, Ut);
    k_bias_init<<<256, 256, 0, stream>>>(bW, bU, bias, h0, bar);

    dim3 g1(H_ / 64, B_ / 16, S_ * 4);
    k_xproj<<<g1, 128, 0, stream>>>(xbf, Wt, xproj);

    float* hlast = out + (size_t)B_ * S_ * H_;
    float* clast = hlast + (size_t)B_ * H_;
    dim3 g2(H_ / 16, B_ / 16);   // 256 blocks, guaranteed co-resident
    k_lstm_scan<<<g2, 128, 0, stream>>>(h0, h1, Ut, xproj, bias, out,
                                        hlast, clast, bar);
}
